// ASAAttention_26903675142286
// MI455X (gfx1250) — compile-verified, matrix operand order checked by reading, not run on hardware
//
#include <hip/hip_runtime.h>

// ---------------------------------------------------------------------------
// ASA attention for MI455X (gfx1250, wave32, WMMA).
//   B=2, S=2048, D_MODEL=768, H=12, D_HEAD=64, F=64.
// Compute-bound (~39 GFLOP on ~25MB of data, all L2-resident at 192MB):
// all GEMMs routed through v_wmma_f32_16x16x32_bf16; attention is flash-style
// (scores never leave VGPRs); bonding mask fused into the score tile.
// V is stored transposed [B,H,64,S] so every WMMA operand load in the
// attention loop is a row-major gather -> clause-paired global_load_b128.
// Q is pre-scaled by 1/sqrt(64)=2^-3 at projection time (exact in bf16).
// ---------------------------------------------------------------------------

#define B_  2
#define S_  2048
#define H_  12
#define DM_ 768
#define DH_ 64
#define F_  64

typedef __attribute__((ext_vector_type(16))) __bf16       v16bf;
typedef __attribute__((ext_vector_type(8)))  float        v8f;
typedef __attribute__((ext_vector_type(8)))  unsigned int v8u;

// ---- POS compatibility matrix as 17 bitmask rows (built at compile time) ---
struct PosRows { unsigned r[17]; };
static constexpr PosRows makePosRows() {
  PosRows P{};
  // (dep,head) index pairs per reference tag order:
  // NOUN0 VERB1 ADJ2 ADV3 PRON4 PROPN5 DET6 ADP7 AUX8 CCONJ9 SCONJ10 NUM11
  // PART12 INTJ13 PUNCT14 SYM15 X16
  const int prs[33][2] = {
    {0,1},{5,1},{4,1},{0,2},{5,2},{4,2},{1,1},{2,0},{2,5},{6,0},{6,5},{11,0},
    {7,0},{7,5},{7,4},{0,0},{5,0},{0,5},{5,5},{3,1},{3,2},{3,3},{8,1},{10,1},
    {8,2},{8,0},{9,0},{9,1},{9,2},{9,3},{9,5},{4,0},{4,5}};
  for (int t = 0; t < 33; ++t) {
    int a = prs[t][0], b = prs[t][1];
    P.r[a] |= 1u << b;  P.r[b] |= 1u << a;
  }
  for (int i = 0; i < 17; ++i) P.r[i] |= (1u << i) | (1u << 14); // diag + PUNCT col
  P.r[14] = 0x1FFFFu;                                           // PUNCT row
  return P;
}
__constant__ PosRows g_pos = makePosRows();

// ---- helpers ---------------------------------------------------------------
__device__ __forceinline__ unsigned short f2bf(float f) {
  unsigned u = __builtin_bit_cast(unsigned, f);
  u += 0x7FFFu + ((u >> 16) & 1u);          // round to nearest even
  return (unsigned short)(u >> 16);
}

__device__ __forceinline__ v8f wmma_bf16(v16bf a, v16bf b, v8f c) {
  return __builtin_amdgcn_wmma_f32_16x16x32_bf16(
      /*neg_a=*/false, a, /*neg_b=*/false, b,
      /*c_mod=*/(short)0, c, /*reuse_a=*/false, /*reuse_b=*/false);
}

// A/B fragment, source row-major in bf16.  ISA 16-bit A 16x32 layout:
// lane L -> row M = L%16; half = L/16; VGPR v holds K pair
//   k = kbase + (v<4 ? 0 : 16) + half*8 + (v&3)*2.
// B mirrored with lane = N (column), gathering K along the source row.
// Compiler coalesces this into two global_load_b128 per fragment.
__device__ __forceinline__ v16bf load_rm(const unsigned short* row, int kbase, int half) {
  v8u t;
#pragma unroll
  for (int v = 0; v < 8; ++v) {
    int k = kbase + ((v < 4) ? 0 : 16) + half * 8 + (v & 3) * 2;
    t[v] = *(const unsigned*)(row + k);     // aligned: k even, rows even-strided
  }
  return __builtin_bit_cast(v16bf, t);
}

__device__ __forceinline__ float redmax16(float v) {
#pragma unroll
  for (int off = 1; off < 16; off <<= 1) v = fmaxf(v, __shfl_xor(v, off, 16));
  return v;
}
__device__ __forceinline__ float redsum16(float v) {
#pragma unroll
  for (int off = 1; off < 16; off <<= 1) v += __shfl_xor(v, off, 16);
  return v;
}

// ---- prep: f32 -> bf16 bulk convert ---------------------------------------
__global__ void cvt_bf16_kernel(const float* __restrict__ s,
                                unsigned short* __restrict__ d, int n) {
  for (int i = blockIdx.x * blockDim.x + threadIdx.x; i < n;
       i += gridDim.x * blockDim.x)
    d[i] = f2bf(s[i]);
}

// req_count[b,s] = sum_f requirements  (f32, exact vs reference)
__global__ void reqcount_kernel(const float* __restrict__ req,
                                float* __restrict__ rc) {
  int t = blockIdx.x * blockDim.x + threadIdx.x;
  if (t >= B_ * S_) return;
  const float* r = req + (long)t * F_;
  float s = 0.f;
#pragma unroll
  for (int f = 0; f < F_; ++f) s += r[f];
  rc[t] = s;
}

// ---- QKV projection: one wave = one 16x16 tile of (x @ W^T + b) ------------
// Q pre-scaled by 0.125 (exact power-of-two in bf16); V stored transposed.
__global__ __launch_bounds__(32) void proj_qkv_kernel(
    const unsigned short* __restrict__ xb,
    const unsigned short* __restrict__ Wqb, const unsigned short* __restrict__ Wkb,
    const unsigned short* __restrict__ Wvb,
    const float* __restrict__ bq, const float* __restrict__ bk,
    const float* __restrict__ bv,
    unsigned short* __restrict__ Qb, unsigned short* __restrict__ Kb,
    unsigned short* __restrict__ Vt) {
  int lane = threadIdx.x, lm = lane & 15, half = lane >> 4;
  int rt = blockIdx.x, ct = blockIdx.y, which = blockIdx.z;
  const unsigned short* W = (which == 0) ? Wqb : (which == 1) ? Wkb : Wvb;
  const float* bias        = (which == 0) ? bq  : (which == 1) ? bk  : bv;

  const unsigned short* xrow = xb + (long)(rt * 16 + lm) * DM_;   // A: lane = M row
  const unsigned short* wrow = W  + (long)(ct * 16 + lm) * DM_;   // B: lane = N col

  v8f acc;
#pragma unroll
  for (int i = 0; i < 8; ++i) acc[i] = 0.f;
  for (int kb = 0; kb < DM_; kb += 32)
    acc = wmma_bf16(load_rm(xrow, kb, half), load_rm(wrow, kb, half), acc);

  int c = ct * 16 + lm, h = c >> 6, d = c & 63;
  float bc = bias[c];
  float scl = (which == 0) ? 0.125f : 1.f;   // fold 1/sqrt(D_HEAD) into Q
#pragma unroll
  for (int r = 0; r < 8; ++r) {
    int mm = r + 8 * half;
    int tok = rt * 16 + mm, b = tok >> 11, s = tok & (S_ - 1);
    unsigned short val = f2bf((acc[r] + bc) * scl);
    long bh = (long)(b * H_ + h);
    if (which == 2)
      Vt[(bh * DH_ + d) * S_ + s] = val;                 // transposed [B,H,64,S]
    else
      ((which == 0) ? Qb : Kb)[(bh * S_ + s) * DH_ + d] = val;
  }
}

// ---- flash attention with fused bonding mask -------------------------------
__global__ __launch_bounds__(32) void attn_kernel(
    const unsigned short* __restrict__ Qb, const unsigned short* __restrict__ Kb,
    const unsigned short* __restrict__ Vt,   // [B,H,64,S]
    const unsigned short* __restrict__ Fb,   // features bf16 [B,S,64]
    const unsigned short* __restrict__ Rb,   // requirements bf16 [B,S,64]
    const int* __restrict__ pos_ids, const float* __restrict__ reqcnt,
    unsigned short* __restrict__ attn) {
  __shared__ unsigned short Plds[16 * 32];   // P tile bounce (C-layout -> A-layout)
  const float NEG = -__builtin_inff();
  int lane = threadIdx.x, lm = lane & 15, half = lane >> 4;
  int qt = blockIdx.x, h = blockIdx.y, b = blockIdx.z;
  int qbase = qt * 16, qend = qbase + 15;
  long bh = (long)(b * H_ + h);

  const unsigned short* Qrow = Qb + (bh * S_ + qbase + lm) * DH_;
  const unsigned short* Rrow = Rb + ((long)b * S_ + qbase + lm) * F_;
  v16bf aQ0 = load_rm(Qrow, 0, half), aQ1 = load_rm(Qrow, 32, half);
  v16bf aR0 = load_rm(Rrow, 0, half), aR1 = load_rm(Rrow, 32, half);

  float rc[8], invrc[8]; int posi[8]; bool pron[8], hasreq[8];
#pragma unroll
  for (int r = 0; r < 8; ++r) {
    int i = qbase + r + 8 * half;
    rc[r]     = reqcnt[b * S_ + i];
    invrc[r]  = 1.f / (rc[r] + 1e-6f);       // hoisted out of the key loop
    hasreq[r] = (rc[r] > 0.f);
    posi[r]   = pos_ids[b * S_ + i];
    pron[r]   = (posi[r] == 4);
  }

  float mrow[8], lrow[8];
  v8f O0, O1, O2, O3;
#pragma unroll
  for (int r = 0; r < 8; ++r) {
    mrow[r] = NEG; lrow[r] = 0.f;
    O0[r] = 0.f; O1[r] = 0.f; O2[r] = 0.f; O3[r] = 0.f;
  }

  for (int jb = 0; jb <= qend; jb += 32) {
    if (jb + 32 <= qend)   // prefetch next K block (global_prefetch_b8)
      __builtin_prefetch(Kb + (bh * S_ + jb + 32 + lm) * DH_, 0, 3);

    float sc[2][8];
#pragma unroll
    for (int t = 0; t < 2; ++t) {
      int j = jb + 16 * t + lm;               // key column for this lane
      const unsigned short* Krow = Kb + (bh * S_ + j) * DH_;
      v8f s;
#pragma unroll
      for (int i = 0; i < 8; ++i) s[i] = 0.f;
      s = wmma_bf16(aQ0, load_rm(Krow, 0, half), s);
      s = wmma_bf16(aQ1, load_rm(Krow, 32, half), s);

      const unsigned short* Frow = Fb + ((long)b * S_ + j) * F_;
      v8f sat;
#pragma unroll
      for (int i = 0; i < 8; ++i) sat[i] = 0.f;
      sat = wmma_bf16(aR0, load_rm(Frow, 0, half), sat);
      sat = wmma_bf16(aR1, load_rm(Frow, 32, half), sat);

      int pj = pos_ids[b * S_ + j];
      bool noun = (pj == 0) || (pj == 5);
#pragma unroll
      for (int r = 0; r < 8; ++r) {
        int i = qbase + r + 8 * half;
        float sv = sat[r];
        bool fully = (sv >= rc[r]) || !hasreq[r];
        bool fblk  = pron[r] && noun && !fully;
        bool posm  = (g_pos.r[posi[r]] >> pj) & 1u;
        bool mask  = (posm && !fblk) || (i == j);
        sc[t][r] = (mask && (j <= i)) ? (s[r] + sv * invrc[r]) : NEG;
      }
    }

    // online softmax update (row stats shared across the 16 lanes of a half)
    float pm0[8], pm1[8];
#pragma unroll
    for (int r = 0; r < 8; ++r) {
      float tmax = redmax16(fmaxf(sc[0][r], sc[1][r]));
      float mnew = fmaxf(mrow[r], tmax);
      float scl  = (mnew > NEG) ? expf(mrow[r] - mnew) : 1.f;
      float p0 = (sc[0][r] > NEG) ? expf(sc[0][r] - mnew) : 0.f;
      float p1 = (sc[1][r] > NEG) ? expf(sc[1][r] - mnew) : 0.f;
      lrow[r] = lrow[r] * scl + redsum16(p0 + p1);
      mrow[r] = mnew;
      O0[r] *= scl; O1[r] *= scl; O2[r] *= scl; O3[r] *= scl;
      pm0[r] = p0; pm1[r] = p1;
    }

    // transpose P through LDS: C-layout element (m = r+8*half, n = lm)
#pragma unroll
    for (int r = 0; r < 8; ++r) {
      int mm = r + 8 * half;
      Plds[mm * 32 + lm]      = f2bf(pm0[r]);
      Plds[mm * 32 + 16 + lm] = f2bf(pm1[r]);
    }
    __syncthreads();
    v16bf aP = load_rm(&Plds[lm * 32], 0, half);   // A-layout read, K = 32 keys
    __syncthreads();

    // V^T rows: lane = output dim column, contiguous K along the row
#pragma unroll
    for (int c = 0; c < 4; ++c) {
      const unsigned short* vtrow = Vt + (bh * DH_ + c * 16 + lm) * S_ + jb;
      v16bf bV = load_rm(vtrow, 0, half);
      v8f* Oc = (c == 0) ? &O0 : (c == 1) ? &O1 : (c == 2) ? &O2 : &O3;
      *Oc = wmma_bf16(aP, bV, *Oc);
    }
  }

  // normalize and store (diagonal guarantees lrow > 0)
#pragma unroll
  for (int r = 0; r < 8; ++r) {
    float inv = (lrow[r] > 0.f) ? 1.f / lrow[r] : 0.f;
    int mm = r + 8 * half;
    unsigned short* dst =
        attn + ((long)(b * S_ + qbase + mm)) * DM_ + h * DH_;
    dst[lm]      = f2bf(O0[r] * inv);
    dst[16 + lm] = f2bf(O1[r] * inv);
    dst[32 + lm] = f2bf(O2[r] * inv);
    dst[48 + lm] = f2bf(O3[r] * inv);
  }
}

// ---- output projection: out = attn @ W_o^T + b_o (f32 out) -----------------
__global__ __launch_bounds__(32) void proj_out_kernel(
    const unsigned short* __restrict__ ab, const unsigned short* __restrict__ Wob,
    const float* __restrict__ bo, float* __restrict__ out) {
  int lane = threadIdx.x, lm = lane & 15, half = lane >> 4;
  int rt = blockIdx.x, ct = blockIdx.y;
  const unsigned short* arow = ab  + (long)(rt * 16 + lm) * DM_;
  const unsigned short* wrow = Wob + (long)(ct * 16 + lm) * DM_;
  v8f acc;
#pragma unroll
  for (int i = 0; i < 8; ++i) acc[i] = 0.f;
  for (int kb = 0; kb < DM_; kb += 32)
    acc = wmma_bf16(load_rm(arow, kb, half), load_rm(wrow, kb, half), acc);
  int c = ct * 16 + lm;
  float bc = bo[c];
#pragma unroll
  for (int r = 0; r < 8; ++r) {
    int mm = r + 8 * half;
    out[(long)(rt * 16 + mm) * DM_ + c] = acc[r] + bc;
  }
}

// ---------------------------------------------------------------------------
extern "C" void kernel_launch(void* const* d_in, const int* in_sizes, int n_in,
                              void* d_out, int out_size, void* d_ws, size_t ws_size,
                              hipStream_t stream) {
  const float* x    = (const float*)d_in[0];
  const float* feat = (const float*)d_in[1];
  const float* req  = (const float*)d_in[2];
  const int*   pos  = (const int*)d_in[3];
  const float* Wq = (const float*)d_in[4];  const float* bq = (const float*)d_in[5];
  const float* Wk = (const float*)d_in[6];  const float* bk = (const float*)d_in[7];
  const float* Wv = (const float*)d_in[8];  const float* bv = (const float*)d_in[9];
  const float* Wo = (const float*)d_in[10]; const float* bo = (const float*)d_in[11];
  float* out = (float*)d_out;

  char* ws = (char*)d_ws;
  size_t off = 0;
  auto take = [&](size_t bytes) -> char* {
    char* p = ws + off;
    off = (off + bytes + 255) & ~(size_t)255;
    return p;
  };
  unsigned short* xb  = (unsigned short*)take((size_t)B_ * S_ * DM_ * 2);
  unsigned short* Wqb = (unsigned short*)take((size_t)DM_ * DM_ * 2);
  unsigned short* Wkb = (unsigned short*)take((size_t)DM_ * DM_ * 2);
  unsigned short* Wvb = (unsigned short*)take((size_t)DM_ * DM_ * 2);
  unsigned short* Wob = (unsigned short*)take((size_t)DM_ * DM_ * 2);
  unsigned short* Fb  = (unsigned short*)take((size_t)B_ * S_ * F_ * 2);
  unsigned short* Rb  = (unsigned short*)take((size_t)B_ * S_ * F_ * 2);
  float*          rc  = (float*)take((size_t)B_ * S_ * 4);
  unsigned short* Qb  = (unsigned short*)take((size_t)B_ * H_ * S_ * DH_ * 2);
  unsigned short* Kb  = (unsigned short*)take((size_t)B_ * H_ * S_ * DH_ * 2);
  unsigned short* Vt  = (unsigned short*)take((size_t)B_ * H_ * S_ * DH_ * 2);
  unsigned short* Ab  = (unsigned short*)take((size_t)B_ * S_ * DM_ * 2);
  (void)ws_size; (void)in_sizes; (void)n_in; (void)out_size;  // ~37.3 MB used

  // 1) convert inputs to bf16
  auto cvt = [&](const float* s, unsigned short* d, int n) {
    int blocks = (n + 255) / 256; if (blocks > 4096) blocks = 4096;
    cvt_bf16_kernel<<<blocks, 256, 0, stream>>>(s, d, n);
  };
  cvt(x,    xb,  B_ * S_ * DM_);
  cvt(Wq,   Wqb, DM_ * DM_);
  cvt(Wk,   Wkb, DM_ * DM_);
  cvt(Wv,   Wvb, DM_ * DM_);
  cvt(Wo,   Wob, DM_ * DM_);
  cvt(feat, Fb,  B_ * S_ * F_);
  cvt(req,  Rb,  B_ * S_ * F_);
  reqcount_kernel<<<(B_ * S_ + 255) / 256, 256, 0, stream>>>(req, rc);

  // 2) Q/K/V projections: 256 row-tiles x 48 col-tiles x {Q,K,V}
  proj_qkv_kernel<<<dim3(B_ * S_ / 16, DM_ / 16, 3), 32, 0, stream>>>(
      xb, Wqb, Wkb, Wvb, bq, bk, bv, Qb, Kb, Vt);

  // 3) flash attention + bonding: (S/16 q-tiles) x H x B waves
  attn_kernel<<<dim3(S_ / 16, H_, B_), 32, 0, stream>>>(
      Qb, Kb, Vt, Fb, Rb, pos, rc, Ab);

  // 4) output projection -> f32 d_out
  proj_out_kernel<<<dim3(B_ * S_ / 16, DM_ / 16), 32, 0, stream>>>(
      Ab, Wob, bo, out);
}
